// MLA_55800215109859
// MI455X (gfx1250) — compile-verified, matrix operand order checked by reading, not run on hardware
//
#include <hip/hip_runtime.h>
#include <hip/hip_fp16.h>

// ---------------------------------------------------------------------------
// MLA (per-token head-vs-head attention) for MI455X / gfx1250.
// Pipeline:
//   prepass : x -> f16, all weights -> f16 transposed (N-major)
//   GEMMs   : v_wmma_f32_16x16x32_f16, 256x128 block tile, BK=32,
//             ASYNC global->LDS double-buffered (global_load_async_to_lds_b128
//             + s_wait_asynccnt), 8 waves x (64x64 C) = 16 WMMA per slab
//   attn    : one wave per token, QK^T + softmax(shfl) + AV, all WMMA
// ---------------------------------------------------------------------------

typedef _Float16 h4   __attribute__((ext_vector_type(4)));
typedef _Float16 h8   __attribute__((ext_vector_type(8)));
typedef _Float16 h16  __attribute__((ext_vector_type(16)));
typedef float    f8   __attribute__((ext_vector_type(8)));

#define CAT16(lo, hi) __builtin_shufflevector(lo, hi, 0,1,2,3,4,5,6,7,8,9,10,11,12,13,14,15)

static __device__ __forceinline__ f8 wmma_f16(h16 a, h16 b, f8 c) {
    return __builtin_amdgcn_wmma_f32_16x16x32_f16(false, a, false, b, (short)0, c, false, false);
}

// async copy 16B global -> LDS (no VGPR staging, tracked by ASYNCcnt)
static __device__ __forceinline__ void async_b128(unsigned lds_addr, const void* gptr) {
    asm volatile("global_load_async_to_lds_b128 %0, %1, off"
                 :: "v"(lds_addr), "v"(gptr) : "memory");
}
#define WAIT_ASYNCCNT(n) asm volatile("s_wait_asynccnt " #n ::: "memory")

// ---------------------------------------------------------------------------
// prepass: elementwise f32 -> f16 (4 elems/thread)
// ---------------------------------------------------------------------------
__global__ __launch_bounds__(256)
void cvt_f16(const float* __restrict__ in, _Float16* __restrict__ out)
{
    const size_t i = ((size_t)blockIdx.x * 256 + threadIdx.x) * 4;
    const float4 f = *(const float4*)(in + i);
    *(h4*)(out + i) = h4{(_Float16)f.x, (_Float16)f.y, (_Float16)f.z, (_Float16)f.w};
}

// prepass: W[K][N] f32  ->  Wt[N][K] f16  (32x32 LDS tile)
__global__ __launch_bounds__(256)
void transpose_cvt(const float* __restrict__ W, _Float16* __restrict__ Wt, int K, int N)
{
    __shared__ _Float16 t[32][33];
    const int tx = threadIdx.x & 31;
    const int ty = threadIdx.x >> 5;           // 0..7
    const int k0 = blockIdx.y * 32;
    const int n0 = blockIdx.x * 32;
#pragma unroll
    for (int r = 0; r < 4; r++)
        t[ty * 4 + r][tx] = (_Float16)W[(size_t)(k0 + ty * 4 + r) * N + n0 + tx];
    __syncthreads();
#pragma unroll
    for (int r = 0; r < 4; r++)
        Wt[(size_t)(n0 + ty * 4 + r) * K + k0 + tx] = t[tx][ty * 4 + r];
}

// ---------------------------------------------------------------------------
// WMMA GEMM (f16 x f16t + f32 bias):
//   C[Mrows x N] = A[Mrows x K] @ Bt[N x K]^T + bias[N]
// 256 threads, C tile 256x128, BK=32, async double-buffered LDS.
// waves 4x2, each wave 64x64 -> 16 accumulators, 16 WMMA per k-slab.
// ---------------------------------------------------------------------------
#define LDH 40   // padded LDS row stride in halfs (20 dwords -> conflict-free)

template<typename TC>
__global__ __launch_bounds__(256)
void wmma_gemm_nt(const _Float16* __restrict__ A, const _Float16* __restrict__ Bt,
                  const float* __restrict__ bias, TC* __restrict__ C,
                  int N, int K)
{
    __shared__ _Float16 As[2][256][LDH];   // 40.0 KB
    __shared__ _Float16 Bs[2][128][LDH];   // 20.0 KB  (total 61.4 KB)

    const int tid  = threadIdx.x;
    const int lane = tid & 31;
    const int wave = tid >> 5;
    const int wm   = wave >> 1;              // 0..3 -> 64-row slab
    const int wn   = wave & 1;               // 0..1 -> 64-col slab

    const size_t bm0 = (size_t)blockIdx.y * 256;
    const int    bn0 = blockIdx.x * 128;

    // staging: 4 threads per row, 8 halfs (16B) each -> 64B rows fully covered
    const int srow = tid >> 2;               // 0..63
    const int scol = (tid & 3) * 8;          // 0..24

    // ISA f16 fragment pattern: lane<16 -> K {0..7,16..23}, lane>=16 -> {8..15,24..31}
    const int fidx = lane & 15;
    const int kb   = (lane >> 4) * 8;

    f8 acc[4][4];
#pragma unroll
    for (int i = 0; i < 4; i++)
#pragma unroll
        for (int j = 0; j < 4; j++)
            acc[i][j] = f8{0.f,0.f,0.f,0.f,0.f,0.f,0.f,0.f};

    // issue one K-slab (6 async b128 per thread: 4 x A, 2 x B)
    auto issueSlab = [&](int k0, int buf) {
#pragma unroll
        for (int c = 0; c < 4; c++) {
            const int row = c * 64 + srow;
            async_b128((unsigned)(size_t)&As[buf][row][scol],
                       &A[(bm0 + row) * K + k0 + scol]);
        }
#pragma unroll
        for (int c = 0; c < 2; c++) {
            const int row = c * 64 + srow;
            async_b128((unsigned)(size_t)&Bs[buf][row][scol],
                       &Bt[(size_t)(bn0 + row) * K + k0 + scol]);
        }
    };

    const int nk = K >> 5;                   // K / 32
    issueSlab(0, 0);

    for (int kt = 0; kt < nk; kt++) {
        const int  cur  = kt & 1;
        const bool more = (kt + 1) < nk;
        if (more) {
            issueSlab((kt + 1) << 5, cur ^ 1);   // prefetch next slab into other buffer
            WAIT_ASYNCCNT(6);                     // slab kt complete (in-order), next may fly
        } else {
            WAIT_ASYNCCNT(0);
        }
        __syncthreads();                          // publish all waves' async writes

        h16 afr[4];
#pragma unroll
        for (int i = 0; i < 4; i++) {
            const int r = wm * 64 + i * 16 + fidx;
            afr[i] = CAT16(*(const h8*)&As[cur][r][kb],
                           *(const h8*)&As[cur][r][kb + 16]);
        }
#pragma unroll
        for (int j = 0; j < 4; j++) {
            const int c = wn * 64 + j * 16 + fidx;
            h16 bfr = CAT16(*(const h8*)&Bs[cur][c][kb],
                            *(const h8*)&Bs[cur][c][kb + 16]);
#pragma unroll
            for (int i = 0; i < 4; i++)
                acc[i][j] = wmma_f16(afr[i], bfr, acc[i][j]);
        }
        __syncthreads();                          // buffer reusable for slab kt+2
    }

    // epilogue: D layout -> lane = col (lane&15), VGPR r = row r (+8 hi half-wave)
    const int rh = (lane >> 4) * 8;
#pragma unroll
    for (int i = 0; i < 4; i++)
#pragma unroll
        for (int j = 0; j < 4; j++) {
            const int    col  = bn0 + wn * 64 + j * 16 + fidx;
            const size_t row0 = bm0 + wm * 64 + i * 16 + rh;
            const float  bv   = bias[col];
#pragma unroll
            for (int r = 0; r < 8; r++)
                C[(row0 + r) * N + col] = (TC)(acc[i][j][r] + bv);
        }
}

// ---------------------------------------------------------------------------
// Fused per-token attention: one wave per token.
//   sim  = q[16,128] @ k[16,128]^T * D^-0.5   (4x wmma, K=128)
//   attn = softmax over head axis (16-lane shfl reduction)
//   out  = attn[16,16] @ v[16,128]            (8x wmma, K padded 16->32)
// ---------------------------------------------------------------------------
__global__ __launch_bounds__(256)
void mla_attention(const _Float16* __restrict__ Q, const _Float16* __restrict__ Kh,
                   const _Float16* __restrict__ V, _Float16* __restrict__ O)
{
    __shared__ _Float16 attnS[8][16][16];
    __shared__ _Float16 vT[8][128][24];      // v transposed per wave: vT[d][g]

    const int tid  = threadIdx.x;
    const int lane = tid & 31;
    const int wave = tid >> 5;
    const size_t tok = (size_t)blockIdx.x * 8 + wave;

    const _Float16* q = Q  + tok * 2048;
    const _Float16* k = Kh + tok * 2048;
    const _Float16* v = V  + tok * 2048;
    _Float16*       o = O  + tok * 2048;

    const int idx = lane & 15;
    const int kb  = (lane >> 4) * 8;
    const int rhi = (lane >> 4) * 8;

    f8 c = f8{0.f,0.f,0.f,0.f,0.f,0.f,0.f,0.f};
#pragma unroll
    for (int ks = 0; ks < 128; ks += 32) {
        h8 alo = *(const h8*)&q[idx * 128 + ks + kb];
        h8 ahi = *(const h8*)&q[idx * 128 + ks + kb + 16];
        h8 blo = *(const h8*)&k[idx * 128 + ks + kb];
        h8 bhi = *(const h8*)&k[idx * 128 + ks + kb + 16];
        c = wmma_f16(CAT16(alo, ahi), CAT16(blo, bhi), c);
    }

    const float scale = 0.08838834764831844f;   // 1/sqrt(128)
    float p[8];
#pragma unroll
    for (int r = 0; r < 8; r++) {
        float x = c[r] * scale;
        float m = x;
#pragma unroll
        for (int msk = 1; msk < 16; msk <<= 1) m = fmaxf(m, __shfl_xor(m, msk));
        const float e = __expf(x - m);
        float s = e;
#pragma unroll
        for (int msk = 1; msk < 16; msk <<= 1) s += __shfl_xor(s, msk);
        p[r] = e / s;
    }

#pragma unroll
    for (int r = 0; r < 8; r++)
        attnS[wave][rhi + r][idx] = (_Float16)p[r];
    {
        const int g  = idx;
        const int d0 = (lane >> 4) * 64;
#pragma unroll 8
        for (int d = 0; d < 64; d++)
            vT[wave][d0 + d][g] = v[g * 128 + d0 + d];
    }
    __syncthreads();

    const h8 z8 = h8{(_Float16)0,(_Float16)0,(_Float16)0,(_Float16)0,
                     (_Float16)0,(_Float16)0,(_Float16)0,(_Float16)0};
    h16 af;
    {
        h8 alo = *(const h8*)&attnS[wave][idx][kb];
        af = CAT16(alo, z8);
    }
#pragma unroll
    for (int nt = 0; nt < 8; nt++) {
        const int d = nt * 16 + idx;
        h16 bf = CAT16(*(const h8*)&vT[wave][d][kb], z8);
        f8 oa = f8{0.f,0.f,0.f,0.f,0.f,0.f,0.f,0.f};
        oa = wmma_f16(af, bf, oa);
#pragma unroll
        for (int r = 0; r < 8; r++)
            o[(rhi + r) * 128 + d] = (_Float16)oa[r];
    }
}

// ---------------------------------------------------------------------------
extern "C" void kernel_launch(void* const* d_in, const int* in_sizes, int n_in,
                              void* d_out, int out_size, void* d_ws, size_t ws_size,
                              hipStream_t stream)
{
    const float* x    = (const float*)d_in[0];
    const float* Wd_k = (const float*)d_in[1];
    const float* bd_k = (const float*)d_in[2];
    const float* Wu_k = (const float*)d_in[3];
    const float* bu_k = (const float*)d_in[4];
    const float* Wd_v = (const float*)d_in[5];
    const float* bd_v = (const float*)d_in[6];
    const float* Wu_v = (const float*)d_in[7];
    const float* bu_v = (const float*)d_in[8];
    const float* Wq   = (const float*)d_in[9];
    const float* bq   = (const float*)d_in[10];
    const float* Wo   = (const float*)d_in[11];
    const float* bo   = (const float*)d_in[12];
    float* out = (float*)d_out;

    const int BS = 4 * 4096;     // 16384 tokens
    const int M  = 2048;
    const int L  = 512;
    const int HD = 2048;         // 16 heads * 128

    // workspace layout (all f16)
    char* ws = (char*)d_ws;
    size_t off = 0;
    _Float16* ck   = (_Float16*)(ws + off); off += (size_t)BS * L  * 2;
    _Float16* cv   = (_Float16*)(ws + off); off += (size_t)BS * L  * 2;
    _Float16* kh   = (_Float16*)(ws + off); off += (size_t)BS * HD * 2;
    _Float16* vh   = (_Float16*)(ws + off); off += (size_t)BS * HD * 2;
    _Float16* qh   = (_Float16*)(ws + off); off += (size_t)BS * HD * 2;
    _Float16* ao   = (_Float16*)(ws + off); off += (size_t)BS * HD * 2;
    _Float16* xh   = (_Float16*)(ws + off); off += (size_t)BS * M  * 2;
    _Float16* tWdk = (_Float16*)(ws + off); off += (size_t)M  * L  * 2;
    _Float16* tWdv = (_Float16*)(ws + off); off += (size_t)M  * L  * 2;
    _Float16* tWuk = (_Float16*)(ws + off); off += (size_t)L  * HD * 2;
    _Float16* tWuv = (_Float16*)(ws + off); off += (size_t)L  * HD * 2;
    _Float16* tWq  = (_Float16*)(ws + off); off += (size_t)M  * HD * 2;
    _Float16* tWo  = (_Float16*)(ws + off); off += (size_t)HD * M  * 2;

    const dim3 blk(256);

    // ---- prepass: x -> f16; weights -> f16 transposed (N-major)
    cvt_f16<<<(size_t)BS * M / 1024, blk, 0, stream>>>(x, xh);
    transpose_cvt<<<dim3(L  / 32, M  / 32), blk, 0, stream>>>(Wd_k, tWdk, M,  L);
    transpose_cvt<<<dim3(L  / 32, M  / 32), blk, 0, stream>>>(Wd_v, tWdv, M,  L);
    transpose_cvt<<<dim3(HD / 32, L  / 32), blk, 0, stream>>>(Wu_k, tWuk, L,  HD);
    transpose_cvt<<<dim3(HD / 32, L  / 32), blk, 0, stream>>>(Wu_v, tWuv, L,  HD);
    transpose_cvt<<<dim3(HD / 32, M  / 32), blk, 0, stream>>>(Wq,   tWq,  M,  HD);
    transpose_cvt<<<dim3(M  / 32, HD / 32), blk, 0, stream>>>(Wo,   tWo,  HD, M);

    const dim3 gDown(L  / 128, BS / 256);   // (4, 64)
    const dim3 gWide(HD / 128, BS / 256);   // (16, 64)

    // down projections: cache = x @ Wd + bd        [BS x 512], K=2048
    wmma_gemm_nt<_Float16><<<gDown, blk, 0, stream>>>(xh, tWdk, bd_k, ck, L, M);
    wmma_gemm_nt<_Float16><<<gDown, blk, 0, stream>>>(xh, tWdv, bd_v, cv, L, M);
    // up projections: k/v = cache @ Wu + bu        [BS x 2048], K=512
    wmma_gemm_nt<_Float16><<<gWide, blk, 0, stream>>>(ck, tWuk, bu_k, kh, HD, L);
    wmma_gemm_nt<_Float16><<<gWide, blk, 0, stream>>>(cv, tWuv, bu_v, vh, HD, L);
    // q = x @ Wq + bq                              [BS x 2048], K=2048
    wmma_gemm_nt<_Float16><<<gWide, blk, 0, stream>>>(xh, tWq, bq, qh, HD, M);
    // per-token attention                          [BS x 2048]
    mla_attention<<<BS / 8, blk, 0, stream>>>(qh, kh, vh, ao);
    // out = attn_out @ Wo + bo                     [BS x 2048] fp32
    wmma_gemm_nt<float><<<gWide, blk, 0, stream>>>(ao, tWo, bo, out, M, HD);
}
